// SuperGlueMultiHeadAttention_45896020525737
// MI455X (gfx1250) — matrix operand order repaired and checked by reading, not compiled, not run on hardware
//
#include <hip/hip_runtime.h>
#include <hip/hip_bf16.h>

// ---------------------------------------------------------------------------
// SuperGlue multi-head attention, CDNA5 (gfx1250) wave32 WMMA implementation.
//   B=8, C=256, H=4, D=64, N=M=2048
// Compute-bound (~2850 FLOP/byte with fused flash attention) -> bf16 WMMA
// (v_wmma_f32_16x16x32_bf16, fp32 accumulate) for every GEMM, async
// global->LDS staging (GLOBAL_LOAD_ASYNC_TO_LDS_B128 + s_wait_asynccnt)
// for the K/V tiles shared by all waves of a workgroup.
// ---------------------------------------------------------------------------

#define Bb 8
#define Cc 256
#define Hh 4
#define Dd 64
#define Nn 2048

typedef __attribute__((ext_vector_type(16))) __bf16  v16bf;
typedef __attribute__((ext_vector_type(2)))  __bf16  v2bf;
typedef __attribute__((ext_vector_type(8)))  float   v8f;
typedef __attribute__((ext_vector_type(4)))  unsigned int v4u;

union FragBF {
    v16bf v;
    v4u   u[2];
    unsigned int   w[8];
    unsigned short s[16];
};

union FragC {
    v8f   v;
    float f[8];
};

// ---- f32 -> bf16 via native fptrunc (backend selects the HW convert) ------
__device__ __forceinline__ unsigned int pack_bf16(float lo, float hi) {
    v2bf t;
    t.x = (__bf16)lo;
    t.y = (__bf16)hi;
    return __builtin_bit_cast(unsigned int, t);
}

__device__ __forceinline__ unsigned short f32_to_bf16(float f) {
    return __builtin_bit_cast(unsigned short, (__bf16)f);
}

// ---- async global -> LDS copy (ASYNCcnt-tracked, per ISA 10.7/15.18) ------
// Generic LDS pointers carry the LDS byte offset in their low 32 bits
// which is what the VDST operand wants.
__device__ __forceinline__ void async_load_b128(const void* lds_ptr, const void* gptr) {
    unsigned int lds_off = (unsigned int)(unsigned long long)lds_ptr;
    unsigned long long ga = (unsigned long long)gptr;
    asm volatile("global_load_async_to_lds_b128 %0, %1, off"
                 :: "v"(lds_off), "v"(ga) : "memory");
}

// ---------------------------------------------------------------------------
// Phase 1: fused QKV projections.
//   proj[b,o,n] = sum_c W[o,c] * X[b,c,n] + bias[o]; channel o = d*H + h.
//   mode 0 -> Qnd[b,h,n,d], mode 1 -> Kmd[b,h,m,d], mode 2 -> Vdm[b,h,d,m]
// One wave computes a 16(o) x 16(n) tile; X tile is transposed through LDS.
// ---------------------------------------------------------------------------
__global__ __launch_bounds__(32) void sg_proj(
    const float* __restrict__ xq, const float* __restrict__ xk, const float* __restrict__ xv,
    const float* __restrict__ wq, const float* __restrict__ bq,
    const float* __restrict__ wk, const float* __restrict__ bk,
    const float* __restrict__ wv, const float* __restrict__ bv,
    unsigned short* __restrict__ Qnd,
    unsigned short* __restrict__ Kmd,
    unsigned short* __restrict__ Vdm)
{
    const int mode = blockIdx.z;
    const float* X;  const float* W;  const float* Bi;
    if (mode == 0)      { X = xq; W = wq; Bi = bq; }
    else if (mode == 1) { X = xk; W = wk; Bi = bk; }
    else                { X = xv; W = wv; Bi = bv; }

    const int o0   = blockIdx.x * 16;
    const int j0   = blockIdx.y * 16;     // position over B*N (N % 16 == 0)
    const int b    = j0 / Nn;
    const int n0   = j0 % Nn;
    const int l    = threadIdx.x;
    const int half = l >> 4;
    const int lid  = l & 15;
    const int kbA  = half * 8;            // A-fragment K base (ISA layout)
    const int kbB  = half * 8;            // B-fragment K base (ISA layout)

    __shared__ unsigned short ldsX[16][32];   // [n_local][c_local], bf16

    FragC acc;
#pragma unroll
    for (int i = 0; i < 8; ++i) acc.f[i] = 0.0f;

    const float* Xb = X + (size_t)b * Cc * Nn;

    for (int k0 = 0; k0 < Cc; k0 += 32) {
        // Stage + transpose X tile: lane loads row c = k0+l, cols n0..n0+15.
        {
            const float4* s4 = (const float4*)(Xb + (size_t)(k0 + l) * Nn + n0);
#pragma unroll
            for (int i = 0; i < 4; ++i) {
                float4 t = s4[i];
                unsigned int p01 = pack_bf16(t.x, t.y);
                unsigned int p23 = pack_bf16(t.z, t.w);
                ldsX[4 * i + 0][l] = (unsigned short)(p01 & 0xffffu);
                ldsX[4 * i + 1][l] = (unsigned short)(p01 >> 16);
                ldsX[4 * i + 2][l] = (unsigned short)(p23 & 0xffffu);
                ldsX[4 * i + 3][l] = (unsigned short)(p23 >> 16);
            }
        }
        // A fragment from W (row o0+lid, K chunks at kbA and kbA+16).
        FragBF a;
        {
            const float* wrow = W + (size_t)(o0 + lid) * Cc + k0 + kbA;
#pragma unroll
            for (int j = 0; j < 4; ++j) a.w[j]     = pack_bf16(wrow[2 * j],      wrow[2 * j + 1]);
#pragma unroll
            for (int j = 0; j < 4; ++j) a.w[4 + j] = pack_bf16(wrow[16 + 2 * j], wrow[16 + 2 * j + 1]);
        }
        // B fragment from LDS (col n = lid, K chunks at kbB and kbB+16).
        FragBF bf_;
        {
            const v4u* p = (const v4u*)&ldsX[lid][kbB];
            bf_.u[0] = p[0];
            bf_.u[1] = p[2];
        }
        acc.v = __builtin_amdgcn_wmma_f32_16x16x32_bf16(
            false, a.v, false, bf_.v, (short)0, acc.v, false, false);
    }

    // Epilogue: bias add, bf16 store into attention-friendly layouts.
#pragma unroll
    for (int r = 0; r < 8; ++r) {
        const int o = o0 + half * 8 + r;
        const int n = n0 + lid;
        const float val = acc.f[r] + Bi[o];
        const unsigned short bvv = f32_to_bf16(val);
        const int d = o >> 2;            // C viewed as (D, H): o = d*H + h
        const int h = o & 3;
        const size_t bh = (size_t)b * Hh + h;
        if (mode == 2) {
            Vdm[(bh * Dd + d) * Nn + n] = bvv;
        } else if (mode == 0) {
            Qnd[(bh * Nn + n) * Dd + d] = bvv;
        } else {
            Kmd[(bh * Nn + n) * Dd + d] = bvv;
        }
    }
}

// ---------------------------------------------------------------------------
// Phase 2: flash attention per (b,h). 8 waves per WG cover 128 query rows;
// K/V m-tiles (32 keys) are staged into LDS with double-buffered
// GLOBAL_LOAD_ASYNC_TO_LDS_B128 (one 16B transfer per thread per tile),
// paced with s_wait_asynccnt + workgroup barriers, and shared by all waves.
// Online softmax; P goes through per-wave LDS for C->A layout conversion.
// Output: AttT[b, n, d*H + h] in bf16 (contiguous channel for phase 3).
// ---------------------------------------------------------------------------
__global__ __launch_bounds__(256) void sg_attn(
    const unsigned short* __restrict__ Qnd,
    const unsigned short* __restrict__ Kmd,
    const unsigned short* __restrict__ Vdm,
    unsigned short* __restrict__ AttT)
{
    const int bh   = blockIdx.y;               // b*H + h
    const int tid  = threadIdx.x;
    const int wid  = tid >> 5;
    const int l    = tid & 31;
    const int half = l >> 4;
    const int lid  = l & 15;
    const int kbA  = half * 8;
    const int kbB  = half * 8;
    const int n0   = blockIdx.x * 128 + wid * 16;

    const unsigned short* Qb = Qnd + (size_t)bh * Nn * Dd;
    const unsigned short* Kb = Kmd + (size_t)bh * Nn * Dd;
    const unsigned short* Vb = Vdm + (size_t)bh * Dd * Nn;

    __shared__ unsigned short ldsK[2][32][64];   // [buf][m_local][d]   4KB/buf
    __shared__ unsigned short ldsV[2][64][32];   // [buf][d][m_local]   4KB/buf
    __shared__ unsigned short ldsP[8][16][32];   // per-wave P tile     8KB

    // Cooperative async staging of one (K,V) m-tile: 256 threads x 1 B128 each.
    const int kml = tid >> 3, kch = tid & 7;     // K: 32 rows x 8 chunks
    const int vdr = tid >> 2, vch = tid & 3;     // V: 64 rows x 4 chunks

    // Prologue: fill buffer 0 (tile m0 = 0).
    async_load_b128(&ldsK[0][kml][kch * 8], Kb + (size_t)kml * Dd + kch * 8);
    async_load_b128(&ldsV[0][vdr][vch * 8], Vb + (size_t)vdr * Nn + vch * 8);

    // Q fragments (2 K-steps over D=64), resident for the whole m loop.
    FragBF qf[2];
    {
        const unsigned short* qrow = Qb + (size_t)(n0 + lid) * Dd;
#pragma unroll
        for (int ks = 0; ks < 2; ++ks) {
            qf[ks].u[0] = *(const v4u*)(qrow + ks * 32 + kbA);
            qf[ks].u[1] = *(const v4u*)(qrow + ks * 32 + 16 + kbA);
        }
    }

    FragC acc[4];
#pragma unroll
    for (int dt = 0; dt < 4; ++dt)
#pragma unroll
        for (int i = 0; i < 8; ++i) acc[dt].f[i] = 0.0f;

    float mrun[8], lrun[8];
#pragma unroll
    for (int r = 0; r < 8; ++r) { mrun[r] = -1e30f; lrun[r] = 0.0f; }

    const float scale = 0.125f;   // 1/sqrt(64)
    const int   NT    = Nn / 32;  // number of m-tiles

    for (int it = 0; it < NT; ++it) {
        const int cur = it & 1;
        if (it + 1 < NT) {
            // Kick off next tile into the other buffer, then wait for the
            // PREVIOUS tile only (2 transfers just issued stay in flight).
            const int m1 = (it + 1) * 32;
            async_load_b128(&ldsK[cur ^ 1][kml][kch * 8],
                            Kb + (size_t)(m1 + kml) * Dd + kch * 8);
            async_load_b128(&ldsV[cur ^ 1][vdr][vch * 8],
                            Vb + (size_t)vdr * Nn + m1 + vch * 8);
            asm volatile("s_wait_asynccnt 0x2" ::: "memory");
        } else {
            asm volatile("s_wait_asynccnt 0x0" ::: "memory");
        }
        __syncthreads();   // buf[cur] visible to every wave

        // ---- S = Q * K^T (two 16-col subtiles, two K-steps over D) ----
        FragC s0, s1;
#pragma unroll
        for (int i = 0; i < 8; ++i) { s0.f[i] = 0.0f; s1.f[i] = 0.0f; }
#pragma unroll
        for (int ks = 0; ks < 2; ++ks) {
            FragBF kf0, kf1;
            const v4u* kr0 = (const v4u*)&ldsK[cur][lid][ks * 32 + kbB];
            const v4u* kr1 = (const v4u*)&ldsK[cur][16 + lid][ks * 32 + kbB];
            kf0.u[0] = kr0[0];  kf0.u[1] = kr0[2];
            kf1.u[0] = kr1[0];  kf1.u[1] = kr1[2];
            s0.v = __builtin_amdgcn_wmma_f32_16x16x32_bf16(
                false, qf[ks].v, false, kf0.v, (short)0, s0.v, false, false);
            s1.v = __builtin_amdgcn_wmma_f32_16x16x32_bf16(
                false, qf[ks].v, false, kf1.v, (short)0, s1.v, false, false);
        }

        // ---- online softmax (row reductions across 16-lane halves) ----
#pragma unroll
        for (int r = 0; r < 8; ++r) {
            const float a0 = s0.f[r] * scale;
            const float a1 = s1.f[r] * scale;
            float rm = fmaxf(a0, a1);
            rm = fmaxf(rm, __shfl_xor(rm, 1, 32));
            rm = fmaxf(rm, __shfl_xor(rm, 2, 32));
            rm = fmaxf(rm, __shfl_xor(rm, 4, 32));
            rm = fmaxf(rm, __shfl_xor(rm, 8, 32));
            const float mnew = fmaxf(mrun[r], rm);
            const float corr = __expf(mrun[r] - mnew);
            const float p0 = __expf(a0 - mnew);
            const float p1 = __expf(a1 - mnew);
            float rs = p0 + p1;
            rs += __shfl_xor(rs, 1, 32);
            rs += __shfl_xor(rs, 2, 32);
            rs += __shfl_xor(rs, 4, 32);
            rs += __shfl_xor(rs, 8, 32);
            lrun[r] = lrun[r] * corr + rs;
            mrun[r] = mnew;
#pragma unroll
            for (int dt = 0; dt < 4; ++dt) acc[dt].f[r] *= corr;

            const int nrow = half * 8 + r;           // C-layout row
            const unsigned int pp = pack_bf16(p0, p1);
            ldsP[wid][nrow][lid]      = (unsigned short)(pp & 0xffffu);
            ldsP[wid][nrow][16 + lid] = (unsigned short)(pp >> 16);
        }

        // ---- reload P in A-fragment layout (per-wave LDS, in-order DS) ----
        FragBF pf;
        pf.u[0] = *(const v4u*)&ldsP[wid][lid][kbA];
        pf.u[1] = *(const v4u*)&ldsP[wid][lid][16 + kbA];

        // ---- acc += P * V' over 4 d-tiles ----
#pragma unroll
        for (int dt = 0; dt < 4; ++dt) {
            FragBF vf;
            const v4u* vr = (const v4u*)&ldsV[cur][dt * 16 + lid][kbB];
            vf.u[0] = vr[0];
            vf.u[1] = vr[2];
            acc[dt].v = __builtin_amdgcn_wmma_f32_16x16x32_bf16(
                false, pf.v, false, vf.v, (short)0, acc[dt].v, false, false);
        }

        __syncthreads();   // all waves done with buf[cur] before it refills
    }

    // ---- normalize + store AttT[b, n, d*H + h] ----
    const int b = bh >> 2;
    const int h = bh & 3;
#pragma unroll
    for (int r = 0; r < 8; ++r) {
        const float inv = 1.0f / lrun[r];
        const int n = n0 + half * 8 + r;
        const size_t base = ((size_t)b * Nn + n) * Cc + h;
#pragma unroll
        for (int dt = 0; dt < 4; ++dt) {
            const int d = dt * 16 + lid;
            AttT[base + (size_t)d * Hh] = f32_to_bf16(acc[dt].f[r] * inv);
        }
    }
}

// ---------------------------------------------------------------------------
// Phase 3: output projection. out[b,o,n] = sum_c wm[o,c]*AttT[b,n,c] + bm[o].
// One wave per 16x16 (o,n) tile; B fragments read contiguous bf16 channels.
// ---------------------------------------------------------------------------
__global__ __launch_bounds__(32) void sg_outproj(
    const unsigned short* __restrict__ AttT,
    const float* __restrict__ Wm, const float* __restrict__ Bm,
    float* __restrict__ Out)
{
    const int o0   = blockIdx.x * 16;
    const int j0   = blockIdx.y * 16;
    const int b    = j0 / Nn;
    const int n0   = j0 % Nn;
    const int l    = threadIdx.x;
    const int half = l >> 4;
    const int lid  = l & 15;
    const int kbA  = half * 8;
    const int kbB  = half * 8;

    FragC acc;
#pragma unroll
    for (int i = 0; i < 8; ++i) acc.f[i] = 0.0f;

    const unsigned short* arow = AttT + ((size_t)b * Nn + n0 + lid) * Cc;
    const float* wrow = Wm + (size_t)(o0 + lid) * Cc;

    for (int k0 = 0; k0 < Cc; k0 += 32) {
        FragBF a;
        {
            const float* wp = wrow + k0 + kbA;
#pragma unroll
            for (int j = 0; j < 4; ++j) a.w[j]     = pack_bf16(wp[2 * j],      wp[2 * j + 1]);
#pragma unroll
            for (int j = 0; j < 4; ++j) a.w[4 + j] = pack_bf16(wp[16 + 2 * j], wp[16 + 2 * j + 1]);
        }
        FragBF bf_;
        {
            const v4u* bp = (const v4u*)(arow + k0 + kbB);
            bf_.u[0] = bp[0];
            bf_.u[1] = bp[2];
        }
        acc.v = __builtin_amdgcn_wmma_f32_16x16x32_bf16(
            false, a.v, false, bf_.v, (short)0, acc.v, false, false);
    }

#pragma unroll
    for (int r = 0; r < 8; ++r) {
        const int o = o0 + half * 8 + r;
        const int n = n0 + lid;
        Out[((size_t)b * Cc + o) * Nn + n] = acc.f[r] + Bm[o];
    }
}

// ---------------------------------------------------------------------------
extern "C" void kernel_launch(void* const* d_in, const int* in_sizes, int n_in,
                              void* d_out, int out_size, void* d_ws, size_t ws_size,
                              hipStream_t stream) {
    const float* xq = (const float*)d_in[0];
    const float* xk = (const float*)d_in[1];
    const float* xv = (const float*)d_in[2];
    const float* wq = (const float*)d_in[3];
    const float* bq = (const float*)d_in[4];
    const float* bk_unused = nullptr; (void)bk_unused;
    const float* wk = (const float*)d_in[5];
    const float* bk = (const float*)d_in[6];
    const float* wv = (const float*)d_in[7];
    const float* bv = (const float*)d_in[8];
    const float* wm = (const float*)d_in[9];
    const float* bm = (const float*)d_in[10];
    float* out = (float*)d_out;

    // Workspace layout (bf16 tensors, 8 MiB each, 32 MiB total):
    //   Qnd [B,H,N,D] | Kmd [B,H,M,D] | Vdm [B,H,D,M] | AttT [B,N,C]
    char* ws = (char*)d_ws;
    const size_t tile = (size_t)Bb * Hh * Nn * Dd * sizeof(unsigned short);  // 8 MiB
    unsigned short* Qnd  = (unsigned short*)(ws);
    unsigned short* Kmd  = (unsigned short*)(ws + tile);
    unsigned short* Vdm  = (unsigned short*)(ws + 2 * tile);
    unsigned short* AttT = (unsigned short*)(ws + 3 * tile);

    // Phase 1: QKV projections (one wave per 16x16 tile, z = q/k/v).
    {
        dim3 grid(Cc / 16, (Bb * Nn) / 16, 3);
        dim3 block(32);
        hipLaunchKernelGGL(sg_proj, grid, block, 0, stream,
                           xq, xk, xv, wq, bq, wk, bk, wv, bv, Qnd, Kmd, Vdm);
    }
    // Phase 2: flash attention per (b,h), async-staged K/V tiles.
    {
        dim3 grid(Nn / 128, Bb * Hh);
        dim3 block(256);
        hipLaunchKernelGGL(sg_attn, grid, block, 0, stream, Qnd, Kmd, Vdm, AttT);
    }
    // Phase 3: output projection.
    {
        dim3 grid(Cc / 16, (Bb * Nn) / 16);
        dim3 block(32);
        hipLaunchKernelGGL(sg_outproj, grid, block, 0, stream, AttT, wm, bm, out);
    }
}
